// DCnv2_11450382811894
// MI455X (gfx1250) — matrix operand order repaired and checked by reading, not compiled, not run on hardware
//
#include <hip/hip_runtime.h>
#include <math.h>

typedef __attribute__((ext_vector_type(16))) _Float16 v16h;
typedef __attribute__((ext_vector_type(8)))  float    v8f;

constexpr int BB   = 16;
constexpr int C1c  = 128;
constexpr int C2c  = 128;
constexpr int Hc   = 64;
constexpr int Wc   = 64;
constexpr int KRED = C1c * 9;          // 1152
constexpr int NPIX = BB * Hc * Wc;     // 65536
constexpr int HW   = Hc * Wc;          // 4096
constexpr int KSTEPS = KRED / 32;      // 36
constexpr float EPSc = 1e-5f;

// Reduction index permutation is TAP-MAJOR:  K' = r*128 + c   (r = 3x3 tap, c = channel)
// so c = K' & 127, r = K' >> 7 -- no integer division anywhere in the hot loops.

static __device__ __forceinline__ v8f wmma_f16(v16h a, v16h b, v8f c) {
  return __builtin_amdgcn_wmma_f32_16x16x32_f16(false, a, false, b, (short)0, c, false, false);
}

// A-matrix 16x32 f16 k order used here (contiguous, not the ISA table order): lane holds m = lane&15;
// element j -> K = j + 16*(lane>>4).
static __device__ __forceinline__ int a_src_k(int lane, int j) {
  return j + ((lane >> 4) << 4);
}

static __device__ __forceinline__ float siluf(float v) { return v / (1.f + __expf(-v)); }
static __device__ __forceinline__ float sigmf(float v) { return 1.f / (1.f + __expf(-v)); }
static __device__ __forceinline__ int clampi(int v, int lo, int hi) {
  return v < lo ? lo : (v > hi ? hi : v);
}

// ---- One-time weight pack: f32 OIHW -> f16 A-fragment-ready [mblock][kstep][lane][16],
// ---- with the tap-major K permutation baked in.
__global__ __launch_bounds__(256) void k_pack_a(const float* __restrict__ w,
                                                _Float16* __restrict__ dst, int nblk)
{
  int i = blockIdx.x * 256 + threadIdx.x;
  int total = nblk * KSTEPS * 512;
  if (i >= total) return;
  int j    = i & 15;
  int lane = (i >> 4) & 31;
  int ks   = (i >> 9) % KSTEPS;
  int mb   = i / (KSTEPS * 512);
  int m  = mb * 16 + (lane & 15);
  int Kp = ks * 32 + a_src_k(lane, j);     // permuted K
  int c  = Kp & 127, r = Kp >> 7;
  dst[i] = (_Float16)w[m * KRED + c * 9 + r];
}

// Pack concatenated offset(18)+mask(9) weights, zero-padded to M=32
__global__ __launch_bounds__(256) void k_pack_a_om(const float* __restrict__ w_off,
                                                   const float* __restrict__ w_mask,
                                                   _Float16* __restrict__ dst)
{
  int i = blockIdx.x * 256 + threadIdx.x;
  int total = 2 * KSTEPS * 512;
  if (i >= total) return;
  int j    = i & 15;
  int lane = (i >> 4) & 31;
  int ks   = (i >> 9) % KSTEPS;
  int mb   = i / (KSTEPS * 512);
  int m  = mb * 16 + (lane & 15);
  int Kp = ks * 32 + a_src_k(lane, j);
  int c  = Kp & 127, r = Kp >> 7;
  float v = 0.f;
  if (m < 18)      v = w_off[m * KRED + c * 9 + r];
  else if (m < 27) v = w_mask[(m - 18) * KRED + c * 9 + r];
  dst[i] = (_Float16)v;
}

// ---------------- Kernel 1: conv3x3 + BN1 + SiLU -> h (f16), 128x128 tile ----------------
__global__ __launch_bounds__(256) void k_conv1(
    const float* __restrict__ x, const _Float16* __restrict__ wpk,
    const float* __restrict__ g1, const float* __restrict__ b1,
    const float* __restrict__ m1, const float* __restrict__ v1,
    _Float16* __restrict__ h16)
{
  __shared__ __align__(32) _Float16 Bs[2][8 * 512];  // double-buffered, fragment-ready
  __shared__ float scl[C2c], sft[C2c];

  const int tid = threadIdx.x;
  const int wave = tid >> 5;
  const int lane = tid & 31;
  const int pixel0 = blockIdx.x * 128;

  if (tid < C2c) {
    float s = g1[tid] * rsqrtf(v1[tid] + EPSc);
    scl[tid] = s;
    sft[tid] = b1[tid] - m1[tid] * s;
  }
  __syncthreads();

  // Per-thread constant im2col pixel (n = tid & 127), hoisted out of K loop
  const int nS = tid & 127;
  const int kk0 = tid >> 7;            // 0 or 1; kk = kk0 + 2*it
  const int pS = pixel0 + nS;
  const int bS = pS >> 12, yS = (pS >> 6) & 63, xS = pS & 63;
  const int slotS = ((nS >> 4) << 9) + ((nS & 15) << 4);   // frag*512 + lpos*16
  const float* xbase = x + (size_t)bS * C1c * HW;

  v8f acc[8] = {};

  for (int ks = 0; ks < KSTEPS; ++ks) {
    _Float16* B = Bs[ks & 1];
#pragma unroll
    for (int it = 0; it < 16; ++it) {
      int kk = kk0 + (it << 1);
      int kidx = ks * 32 + kk;
      int c = kidx & 127;              // tap-major: channel
      int r = kidx >> 7;               // tap 0..8
      int ky = (r * 11) >> 5;          // r/3 without division
      int kx = r - ky * 3;
      int iy = yS - 1 + ky, ix = xS - 1 + kx;
      bool ok = ((unsigned)iy < (unsigned)Hc) & ((unsigned)ix < (unsigned)Wc);
      float v = xbase[c * HW + (iy & 63) * Wc + (ix & 63)];  // branch-free masked gather
      B[slotS + ((kk >> 4) << 8) + (kk & 15)] = (_Float16)(ok ? v : 0.f);
    }
    __syncthreads();
    v16h a = *((const v16h*)wpk + (wave * KSTEPS + ks) * 32 + lane);
    const v16h* bp = (const v16h*)B + lane;
#pragma unroll
    for (int q = 0; q < 8; ++q)
      acc[q] = wmma_f16(a, bp[q * 32], acc[q]);
  }

  const int mrow = (lane >> 4) << 3;
#pragma unroll
  for (int q = 0; q < 8; ++q) {
    int pixel = pixel0 + q * 16 + (lane & 15);
    int b = pixel >> 12, y = (pixel >> 6) & 63, xx = pixel & 63;
    _Float16* hb = h16 + (size_t)b * C2c * HW + y * Wc + xx;
#pragma unroll
    for (int r = 0; r < 8; ++r) {
      int m = wave * 16 + mrow + r;
      float v = acc[q][r] * scl[m] + sft[m];
      hb[m * HW] = (_Float16)siluf(v);
    }
  }
}

// ---------- Kernel 2: fused offset(18)+mask(9) conv, padded M=32, 32x128 tile ----------
__global__ __launch_bounds__(256) void k_offmask(
    const _Float16* __restrict__ h16, const _Float16* __restrict__ wpk,
    const float* __restrict__ b_off, const float* __restrict__ b_mask,
    float* __restrict__ offb, float* __restrict__ maskb)
{
  __shared__ __align__(32) _Float16 Bs[2][8 * 512];

  const int tid = threadIdx.x;
  const int wave = tid >> 5;
  const int lane = tid & 31;
  const int mi = wave & 1;             // M block
  const int nb0 = (wave >> 1) << 1;    // first of 2 N frags for this wave
  const int pixel0 = blockIdx.x * 128;

  const int nS = tid & 127;
  const int kk0 = tid >> 7;
  const int pS = pixel0 + nS;
  const int bS = pS >> 12, yS = (pS >> 6) & 63, xS = pS & 63;
  const int slotS = ((nS >> 4) << 9) + ((nS & 15) << 4);
  const _Float16* hbase = h16 + (size_t)bS * C2c * HW;

  v8f acc0 = {}, acc1 = {};

  for (int ks = 0; ks < KSTEPS; ++ks) {
    _Float16* B = Bs[ks & 1];
#pragma unroll
    for (int it = 0; it < 16; ++it) {
      int kk = kk0 + (it << 1);
      int kidx = ks * 32 + kk;
      int c = kidx & 127;
      int r = kidx >> 7;
      int ky = (r * 11) >> 5;
      int kx = r - ky * 3;
      int iy = yS - 1 + ky, ix = xS - 1 + kx;
      bool ok = ((unsigned)iy < (unsigned)Hc) & ((unsigned)ix < (unsigned)Wc);
      _Float16 v = hbase[c * HW + (iy & 63) * Wc + (ix & 63)];
      B[slotS + ((kk >> 4) << 8) + (kk & 15)] = ok ? v : (_Float16)0.f;
    }
    __syncthreads();
    v16h a = *((const v16h*)wpk + (mi * KSTEPS + ks) * 32 + lane);
    const v16h* bp = (const v16h*)B + lane;
    acc0 = wmma_f16(a, bp[nb0 * 32],       acc0);
    acc1 = wmma_f16(a, bp[(nb0 + 1) * 32], acc1);
  }

  const int mrow = (lane >> 4) << 3;
#pragma unroll
  for (int e = 0; e < 2; ++e) {
    v8f acc = e ? acc1 : acc0;
    int pixel = pixel0 + (nb0 + e) * 16 + (lane & 15);
    int b = pixel >> 12, y = (pixel >> 6) & 63, xx = pixel & 63;
    int sp = y * Wc + xx;
#pragma unroll
    for (int r = 0; r < 8; ++r) {
      int m = mi * 16 + mrow + r;
      float v = acc[r];
      if (m < 18)
        offb[((size_t)b * 18 + m) * HW + sp] = v + b_off[m];
      else if (m < 27)
        maskb[((size_t)b * 9 + (m - 18)) * HW + sp] = sigmf(v + b_mask[m - 18]);
    }
  }
}

// ---------- Kernel 3: deformable conv + BN2 + SiLU, 128x128 tile ----------
__global__ __launch_bounds__(256) void k_deform(
    const _Float16* __restrict__ h16, const _Float16* __restrict__ wpk,
    const float* __restrict__ offb, const float* __restrict__ maskb,
    const float* __restrict__ b_d,
    const float* __restrict__ g2, const float* __restrict__ b2,
    const float* __restrict__ m2, const float* __restrict__ v2,
    float* __restrict__ out)
{
  __shared__ __align__(32) _Float16 Bs[2][8 * 512];
  __shared__ float scl[C2c], sft[C2c];
  __shared__ int   tO00[1152], tO01[1152], tO10[1152], tO11[1152];
  __shared__ float tW00[1152], tW01[1152], tW10[1152], tW11[1152];

  const int tid = threadIdx.x;
  const int wave = tid >> 5;
  const int lane = tid & 31;
  const int pixel0 = blockIdx.x * 128;

  if (tid < C2c) {
    float s = g2[tid] * rsqrtf(v2[tid] + EPSc);
    scl[tid] = s;
    sft[tid] = b_d[tid] * s + (b2[tid] - m2[tid] * s);  // fold conv bias into BN shift
  }

  // Per-(pixel, tap) sampling table: clamped corner offsets + mask-premultiplied
  // bilinear weights (OOB corners get weight 0).  1152 = 128 pixels * 9 taps.
  for (int i = tid; i < 1152; i += 256) {
    int n = i / 9, k = i - (i / 9) * 9;    // cold path: division OK (runs 4.5x per thread)
    int pixel = pixel0 + n;
    int b = pixel >> 12, y = (pixel >> 6) & 63, xx = pixel & 63;
    int ky = (k * 11) >> 5, kx = k - ky * 3;
    int sp = y * Wc + xx;
    float oy = offb[((size_t)b * 18 + 2 * k    ) * HW + sp];
    float ox = offb[((size_t)b * 18 + 2 * k + 1) * HW + sp];
    float mv = maskb[((size_t)b * 9 + k) * HW + sp];
    float py = (float)(y - 1 + ky) + oy;
    float px = (float)(xx - 1 + kx) + ox;
    float fy = floorf(py), fx = floorf(px);
    float wy = py - fy, wx = px - fx;
    int y0 = (int)fy, x0 = (int)fx;
    int y1 = y0 + 1, x1 = x0 + 1;
    bool vy0 = (unsigned)y0 < (unsigned)Hc, vy1 = (unsigned)y1 < (unsigned)Hc;
    bool vx0 = (unsigned)x0 < (unsigned)Wc, vx1 = (unsigned)x1 < (unsigned)Wc;
    int y0c = clampi(y0, 0, Hc - 1), y1c = clampi(y1, 0, Hc - 1);
    int x0c = clampi(x0, 0, Wc - 1), x1c = clampi(x1, 0, Wc - 1);
    tO00[i] = y0c * Wc + x0c; tO01[i] = y0c * Wc + x1c;
    tO10[i] = y1c * Wc + x0c; tO11[i] = y1c * Wc + x1c;
    tW00[i] = (vy0 && vx0) ? (1.f - wy) * (1.f - wx) * mv : 0.f;
    tW01[i] = (vy0 && vx1) ? (1.f - wy) * wx * mv : 0.f;
    tW10[i] = (vy1 && vx0) ? wy * (1.f - wx) * mv : 0.f;
    tW11[i] = (vy1 && vx1) ? wy * wx * mv : 0.f;
  }
  __syncthreads();   // tables + scl/sft visible before first staging reads

  const int nS = tid & 127;
  const int kk0 = tid >> 7;
  const int pS = pixel0 + nS;
  const int bS = pS >> 12;
  const int slotS = ((nS >> 4) << 9) + ((nS & 15) << 4);
  const _Float16* hS = h16 + (size_t)bS * C2c * HW;

  v8f acc[8] = {};

  for (int ks = 0; ks < KSTEPS; ++ks) {
    _Float16* B = Bs[ks & 1];
#pragma unroll
    for (int it = 0; it < 16; ++it) {
      int kk = kk0 + (it << 1);
      int kidx = ks * 32 + kk;
      int c = kidx & 127;              // tap-major: channel
      int r = kidx >> 7;               // tap 0..8
      int t = nS * 9 + r;
      const _Float16* hp = hS + c * HW;
      float v = tW00[t] * (float)hp[tO00[t]] + tW01[t] * (float)hp[tO01[t]]
              + tW10[t] * (float)hp[tO10[t]] + tW11[t] * (float)hp[tO11[t]];
      B[slotS + ((kk >> 4) << 8) + (kk & 15)] = (_Float16)v;
    }
    __syncthreads();
    v16h a = *((const v16h*)wpk + (wave * KSTEPS + ks) * 32 + lane);
    const v16h* bp = (const v16h*)B + lane;
#pragma unroll
    for (int q = 0; q < 8; ++q)
      acc[q] = wmma_f16(a, bp[q * 32], acc[q]);
  }

  const int mrow = (lane >> 4) << 3;
#pragma unroll
  for (int q = 0; q < 8; ++q) {
    int pixel = pixel0 + q * 16 + (lane & 15);
    int b = pixel >> 12, y = (pixel >> 6) & 63, xx = pixel & 63;
    float* ob = out + (size_t)b * C2c * HW + y * Wc + xx;
#pragma unroll
    for (int r = 0; r < 8; ++r) {
      int m = wave * 16 + mrow + r;
      float v = acc[q][r] * scl[m] + sft[m];
      ob[m * HW] = siluf(v);
    }
  }
}

extern "C" void kernel_launch(void* const* d_in, const int* in_sizes, int n_in,
                              void* d_out, int out_size, void* d_ws, size_t ws_size,
                              hipStream_t stream)
{
  const float* x      = (const float*)d_in[0];
  const float* w1     = (const float*)d_in[1];
  const float* g1     = (const float*)d_in[2];
  const float* b1     = (const float*)d_in[3];
  const float* m1     = (const float*)d_in[4];
  const float* v1     = (const float*)d_in[5];
  const float* w_off  = (const float*)d_in[6];
  const float* b_off  = (const float*)d_in[7];
  const float* w_mask = (const float*)d_in[8];
  const float* b_mask = (const float*)d_in[9];
  const float* w_d    = (const float*)d_in[10];
  const float* b_d    = (const float*)d_in[11];
  const float* g2     = (const float*)d_in[12];
  const float* b2     = (const float*)d_in[13];
  const float* m2     = (const float*)d_in[14];
  const float* v2     = (const float*)d_in[15];
  float* out = (float*)d_out;

  // Workspace partition (all offsets 32B aligned)
  char* ws = (char*)d_ws;
  _Float16* h16   = (_Float16*)ws;                                  // 16 MB
  size_t off0 = (size_t)BB * C2c * HW * sizeof(_Float16);
  float* offb     = (float*)(ws + off0);                            // 4.5 MB
  size_t off1 = off0 + (size_t)BB * 18 * HW * sizeof(float);
  float* maskb    = (float*)(ws + off1);                            // 2.25 MB
  size_t off2 = off1 + (size_t)BB * 9 * HW * sizeof(float);
  _Float16* wpk1  = (_Float16*)(ws + off2);                         // 288 KB
  size_t off3 = off2 + (size_t)C2c * KRED * sizeof(_Float16);
  _Float16* wpkom = (_Float16*)(ws + off3);                         // 72 KB
  size_t off4 = off3 + (size_t)32 * KRED * sizeof(_Float16);
  _Float16* wpkd  = (_Float16*)(ws + off4);                         // 288 KB

  dim3 blk(256);
  k_pack_a   <<<(8 * KSTEPS * 512 + 255) / 256, blk, 0, stream>>>(w1, wpk1, 8);
  k_pack_a_om<<<(2 * KSTEPS * 512 + 255) / 256, blk, 0, stream>>>(w_off, w_mask, wpkom);
  k_pack_a   <<<(8 * KSTEPS * 512 + 255) / 256, blk, 0, stream>>>(w_d, wpkd, 8);

  k_conv1  <<<NPIX / 128, blk, 0, stream>>>(x, wpk1, g1, b1, m1, v1, h16);
  k_offmask<<<NPIX / 128, blk, 0, stream>>>(h16, wpkom, b_off, b_mask, offb, maskb);
  k_deform <<<NPIX / 128, blk, 0, stream>>>(h16, wpkd, offb, maskb, b_d, g2, b2, m2, v2, out);
}
